// EndPointRepr_69750268887124
// MI455X (gfx1250) — matrix operand order repaired and checked by reading, not compiled, not run on hardware
//
#include <hip/hip_runtime.h>

// ---- problem dimensions (from reference) ----
#define BDIM 8
#define SDIM 2048
#define DDIM 1024     // K
#define PDIM 256      // N
#define NQ   8192
#define MDIM (BDIM * SDIM)   // 16384 rows

typedef __attribute__((ext_vector_type(4)))  float  v4f;
typedef __attribute__((ext_vector_type(8)))  float  v8f;
typedef __attribute__((ext_vector_type(8)))  __bf16 v8bf;
typedef __attribute__((ext_vector_type(16))) __bf16 v16bf;
typedef __attribute__((ext_vector_type(4)))  unsigned short v4u16;

// ---- f32 <-> bf16 helpers (RNE) ----
__device__ __forceinline__ unsigned short f32_to_bf16(float x) {
    unsigned int u = __float_as_uint(x);
    u += 0x7FFFu + ((u >> 16) & 1u);   // round to nearest even
    return (unsigned short)(u >> 16);
}
__device__ __forceinline__ float bf16_to_f32(unsigned short h) {
    return __uint_as_float(((unsigned int)h) << 16);
}

// ---------------------------------------------------------------------------
// Kernel 1: split X (f32) into bf16 hi/lo pair:  x = hi + lo  (~16 mantissa
// bits captured). Memory bound: 64 MB read + 64 MB write.
// ---------------------------------------------------------------------------
__global__ void decompose_x_kernel(const float* __restrict__ X,
                                   unsigned short* __restrict__ Xhi,
                                   unsigned short* __restrict__ Xlo) {
    int idx = blockIdx.x * blockDim.x + threadIdx.x;   // 0 .. MDIM*DDIM/4
    v4f x = ((const v4f*)X)[idx];
    v4u16 hi, lo;
#pragma unroll
    for (int i = 0; i < 4; ++i) {
        unsigned short h = f32_to_bf16(x[i]);
        hi[i] = h;
        lo[i] = f32_to_bf16(x[i] - bf16_to_f32(h));
    }
    ((v4u16*)Xhi)[idx] = hi;
    ((v4u16*)Xlo)[idx] = lo;
}

// ---------------------------------------------------------------------------
// Kernel 2: transpose + split W [K,N] -> Wt hi/lo [N,K] (K-contiguous bf16,
// matching the WMMA bf16 B-operand per-lane layout).
// ---------------------------------------------------------------------------
__global__ void transpose_decompose_w_kernel(const float* __restrict__ W,
                                             unsigned short* __restrict__ Wthi,
                                             unsigned short* __restrict__ Wtlo) {
    int idx = blockIdx.x * blockDim.x + threadIdx.x;   // 0 .. DDIM*PDIM
    int k = idx / PDIM;
    int n = idx % PDIM;
    float x = W[idx];
    unsigned short h = f32_to_bf16(x);
    Wthi[n * DDIM + k] = h;
    Wtlo[n * DDIM + k] = f32_to_bf16(x - bf16_to_f32(h));
}

// ---------------------------------------------------------------------------
// Kernel 3: h = X @ W + b via 3xBF16 split GEMM on V_WMMA_F32_16X16X32_BF16:
//   acc += Ahi*Bhi + Ahi*Blo + Alo*Bhi      (f32 accumulate; err ~2^-16)
//
// Block = 256 threads = 8 waves. Wave w: row-tile (blockIdx.x*4 + w/2)
// (16 M rows) and N-half (w&1)*128 (8 tiles of 16 cols). K steps by 32.
//
// A (16x32 bf16) per lane: M = lane&15; contiguous K range [k+lg*16,+16)
// B (32x16 bf16) per lane: N = lane&15; contiguous K range [k+lg*16,+16)
// C/D (f32): VGPR r, lanes 0-15 -> M=r, N=lane; lanes 16-31 -> M=r+8.
// ---------------------------------------------------------------------------
__global__ void __launch_bounds__(256, 1)
gemm_bf16x3_kernel(const unsigned short* __restrict__ Xhi_,
                   const unsigned short* __restrict__ Xlo_,
                   const unsigned short* __restrict__ Wthi_,
                   const unsigned short* __restrict__ Wtlo_,
                   const float* __restrict__ bias,
                   float* __restrict__ H) {
    const int lane = threadIdx.x & 31;
    const int wave = threadIdx.x >> 5;
    const int lm   = lane & 15;
    const int lg   = lane >> 4;          // 0 or 1

    const int rowTile = blockIdx.x * 4 + (wave >> 1);
    const int m0      = rowTile * 16;
    const int n0      = (wave & 1) * 128;

    const __bf16* Xhi  = (const __bf16*)Xhi_;
    const __bf16* Xlo  = (const __bf16*)Xlo_;
    const __bf16* Wthi = (const __bf16*)Wthi_;
    const __bf16* Wtlo = (const __bf16*)Wtlo_;

    const __bf16* pAhi = Xhi  + (size_t)(m0 + lm) * DDIM + lg * 16;
    const __bf16* pAlo = Xlo  + (size_t)(m0 + lm) * DDIM + lg * 16;
    const __bf16* pBhi = Wthi + (size_t)(n0 + lm) * DDIM + lg * 16;
    const __bf16* pBlo = Wtlo + (size_t)(n0 + lm) * DDIM + lg * 16;

    v8f acc[8] = {};

    for (int k = 0; k < DDIM; k += 32) {
        v8bf ah0 = *(const v8bf*)(pAhi + k);
        v8bf ah1 = *(const v8bf*)(pAhi + k + 8);
        v16bf ahi = __builtin_shufflevector(ah0, ah1,
            0,1,2,3,4,5,6,7,8,9,10,11,12,13,14,15);
        v8bf al0 = *(const v8bf*)(pAlo + k);
        v8bf al1 = *(const v8bf*)(pAlo + k + 8);
        v16bf alo = __builtin_shufflevector(al0, al1,
            0,1,2,3,4,5,6,7,8,9,10,11,12,13,14,15);
#pragma unroll
        for (int t = 0; t < 8; ++t) {
            v16bf bhi = *(const v16bf*)(pBhi + (size_t)t * 16 * DDIM + k);
            v16bf blo = *(const v16bf*)(pBlo + (size_t)t * 16 * DDIM + k);
            acc[t] = __builtin_amdgcn_wmma_f32_16x16x32_bf16(
                false, ahi, false, bhi, (short)0, acc[t], false, false);
            acc[t] = __builtin_amdgcn_wmma_f32_16x16x32_bf16(
                false, ahi, false, blo, (short)0, acc[t], false, false);
            acc[t] = __builtin_amdgcn_wmma_f32_16x16x32_bf16(
                false, alo, false, bhi, (short)0, acc[t], false, false);
        }
    }

    // Bias add + store (C/D layout: M = m0 + lg*8 + r, N = n0 + t*16 + lm).
#pragma unroll
    for (int t = 0; t < 8; ++t) {
        const int n  = n0 + t * 16 + lm;
        const float bv = bias[n];
        float* out0 = H + (size_t)(m0 + lg * 8) * PDIM + n;
#pragma unroll
        for (int r = 0; r < 8; ++r) {
            out0[(size_t)r * PDIM] = acc[t][r] + bv;
        }
    }
}

// ---------------------------------------------------------------------------
// Kernel 4: span gather. out = [res1 (NQ x 512) ; res2 (NQ x 512)] flat.
// ---------------------------------------------------------------------------
__global__ void gather_kernel(const float* __restrict__ H,
                              const int* __restrict__ qb,
                              const int* __restrict__ s1,
                              const int* __restrict__ e1,
                              const int* __restrict__ s2,
                              const int* __restrict__ e2,
                              float* __restrict__ out) {
    int idx = blockIdx.x * blockDim.x + threadIdx.x;  // 0 .. 2*NQ*128
    const int half = NQ * 128;
    const int which = (idx >= half) ? 1 : 0;
    const int r  = idx - which * half;
    const int q  = r >> 7;       // / 128
    const int c4 = r & 127;      // float4 column within 512-float row

    const int b = qb[q];
    const int s = which ? s2[q] : s1[q];
    const int e = which ? e2[q] : e1[q];

    v4f val = {};
    if (e >= s) {
        const int src = (c4 < 64) ? s : e;   // first half = start, second = end
        const int cc  = (c4 & 63) * 4;
        val = *(const v4f*)(H + ((size_t)b * SDIM + src) * PDIM + cc);
    }
    *(v4f*)(out + (size_t)idx * 4) = val;
}

// ---------------------------------------------------------------------------
extern "C" void kernel_launch(void* const* d_in, const int* in_sizes, int n_in,
                              void* d_out, int out_size, void* d_ws, size_t ws_size,
                              hipStream_t stream) {
    (void)in_sizes; (void)n_in; (void)out_size; (void)ws_size;
    // setup_inputs order:
    // 0: flag (unused), 1: encoded_input, 2: start_ids_1, 3: end_ids_1,
    // 4: query_batch_idx, 5: start_ids_2, 6: end_ids_2, 7: W, 8: b
    const float* X    = (const float*)d_in[1];
    const int*   s1   = (const int*)d_in[2];
    const int*   e1   = (const int*)d_in[3];
    const int*   qb   = (const int*)d_in[4];
    const int*   s2   = (const int*)d_in[5];
    const int*   e2   = (const int*)d_in[6];
    const float* W    = (const float*)d_in[7];
    const float* bias = (const float*)d_in[8];

    // Workspace layout (bytes):
    //   Xhi : [0, 32MB)        Xlo : [32MB, 64MB)
    //   Wthi: [64MB, +512KB)   Wtlo: [+512KB, +1MB)
    //   H   : [65MB, 81MB)
    char* ws = (char*)d_ws;
    const size_t xBytes = (size_t)MDIM * DDIM * 2;          // 32 MB
    const size_t wBytes = (size_t)PDIM * DDIM * 2;          // 512 KB
    unsigned short* Xhi  = (unsigned short*)(ws);
    unsigned short* Xlo  = (unsigned short*)(ws + xBytes);
    unsigned short* Wthi = (unsigned short*)(ws + 2 * xBytes);
    unsigned short* Wtlo = (unsigned short*)(ws + 2 * xBytes + wBytes);
    float*          Hbuf = (float*)(ws + 2 * xBytes + 2 * wBytes);

    decompose_x_kernel<<<(MDIM * DDIM / 4) / 256, 256, 0, stream>>>(X, Xhi, Xlo);
    transpose_decompose_w_kernel<<<(DDIM * PDIM) / 256, 256, 0, stream>>>(W, Wthi, Wtlo);
    gemm_bf16x3_kernel<<<MDIM / 64, 256, 0, stream>>>(Xhi, Xlo, Wthi, Wtlo, bias, Hbuf);
    gather_kernel<<<(2 * NQ * 128) / 256, 256, 0, stream>>>(
        Hbuf, qb, s1, e1, s2, e2, (float*)d_out);
}
